// Global_Fai_score_76046690943331
// MI455X (gfx1250) — matrix operand order repaired and checked by reading, not compiled, not run on hardware
//
#include <hip/hip_runtime.h>

typedef __attribute__((ext_vector_type(16))) _Float16 v16h;
typedef __attribute__((ext_vector_type(8)))  float    v8f;

#define VSZ 100000
#define DD  300
#define FF  768
#define MM  64
#define NN  512
#define DOC 100
#define BODY 200
#define HH  5
#define HD  60

__device__ inline v8f wmma_f16(v16h a, v16h b, v8f c) {
  return __builtin_amdgcn_wmma_f32_16x16x32_f16(false, a, false, b, (short)0, c, false, false);
}

// ---- WMMA operand loaders (contiguous k = 16*(lane/16) + e for both A and B, wave32) ----
// A 16x32 f16 from row-major [M][K]: per lane one contiguous 16-half run
__device__ inline v16h ldA(const _Float16* S, int ld) {
  int lane = threadIdx.x & 31;
  int m = lane & 15;
  int kb = (lane & 16) ? 16 : 0;
  v16h a;
#pragma unroll
  for (int e = 0; e < 16; ++e) {
    int v = e >> 1, p = e & 1;
    int k = kb + (v << 1) + p;
    a[e] = S[m * ld + k];
  }
  return a;
}
// B 32x16 f16 from N-major source stored [N][K]: per lane one contiguous 16-half run
__device__ inline v16h ldBt(const _Float16* S, int ld) {
  int lane = threadIdx.x & 31;
  int n = lane & 15;
  int kb = (lane & 16) ? 16 : 0;
  v16h b;
#pragma unroll
  for (int e = 0; e < 16; ++e) {
    int v = e >> 1, p = e & 1;
    int k = kb + (v << 1) + p;
    b[e] = S[n * ld + k];
  }
  return b;
}
// store D (f32 8 VGPRs) as f16 with scale: VGPR r -> row (r + 8*(lane>=16)), col lane%16
__device__ inline void stD16(_Float16* S, int ld, v8f c, float scale) {
  int lane = threadIdx.x & 31;
  int n = lane & 15;
  int mb = (lane & 16) ? 8 : 0;
#pragma unroll
  for (int r = 0; r < 8; ++r) S[(mb + r) * ld + n] = (_Float16)(c[r] * scale);
}

// ---- block reductions (blockDim.x == 256) ----
__device__ inline float bsum(float v, float* red) {
  int t = threadIdx.x;
  red[t] = v; __syncthreads();
  for (int s = 128; s > 0; s >>= 1) { if (t < s) red[t] += red[t + s]; __syncthreads(); }
  float r = red[0]; __syncthreads();
  return r;
}
__device__ inline float bmax(float v, float* red) {
  int t = threadIdx.x;
  red[t] = v; __syncthreads();
  for (int s = 128; s > 0; s >>= 1) { if (t < s) red[t] = fmaxf(red[t], red[t + s]); __syncthreads(); }
  float r = red[0]; __syncthreads();
  return r;
}

// ---------------- kernels ----------------

__global__ void k_zero(float* keep) {
  int i = blockIdx.x * 256 + threadIdx.x;
  if (i < NN) keep[i] = 0.f;
}

// one-shot: W_mc [F][D][5] f32 -> Wt [5][F][320] f16 (N-major, zero padded d>=300)
__global__ void k_wmc_t(const float* W, _Float16* Wt) {
  long idx = (long)blockIdx.x * 256 + threadIdx.x;
  if (idx >= (long)5 * FF * 320) return;
  int d = (int)(idx % 320);
  long r = idx / 320;
  int f = (int)(r % FF);
  int k = (int)(r / FF);
  Wt[idx] = (d < DD) ? (_Float16)W[((long)f * DD + d) * 5 + k] : (_Float16)0.f;
}

// ms: conv1d k=2 valid, relu, mean over 3 positions. One block per mention.
__global__ __launch_bounds__(256) void k_conv_ms(const float* embed, const int* men_ids,
                                                 const float* W, const float* b, float* ms) {
  __shared__ float X[4][DD];
  int m = blockIdx.x, tid = threadIdx.x;
  for (int i = tid; i < 4 * 75; i += 256) {
    int t = i / 75, d4 = (i - t * 75) * 4;
    float4 v = *(const float4*)(embed + (long)men_ids[m * 4 + t] * DD + d4);
    X[t][d4 + 0] = v.x; X[t][d4 + 1] = v.y; X[t][d4 + 2] = v.z; X[t][d4 + 3] = v.w;
  }
  __syncthreads();
  for (int f = tid; f < FF; f += 256) {
    float acc = 0.f;
    for (int t = 0; t < 3; ++t) {
      float y = b[f];
      const float* wf = W + (long)f * DD * 2;
      for (int d = 0; d < DD; ++d)
        y += X[t][d] * wf[d * 2 + 0] + X[t + 1][d] * wf[d * 2 + 1];
      acc += fmaxf(y, 0.f);
    }
    ms[m * FF + f] = acc * (1.f / 3.f);
  }
}

// mc: conv1d k=5 valid as 5 shifted WMMA matmuls; B fed straight from pre-transposed
// global Wt (per-lane contiguous, L2 resident). grid (6 f-groups, 64 mentions)
__global__ __launch_bounds__(256) void k_conv_mc(const float* embed, const int* ctx_ids,
                                                 const _Float16* Wt, const float* b, float* mc) {
  __shared__ _Float16 X[52][320];    // ctx tokens, zero padded (32.5KB)
  int fg = blockIdx.x, m = blockIdx.y, tid = threadIdx.x;
  int wave = tid >> 5, lane = tid & 31;
  for (int i = tid; i < 52 * 320; i += 256) ((_Float16*)X)[i] = (_Float16)0.f;
  __syncthreads();
  for (int i = tid; i < 50 * 75; i += 256) {
    int t = i / 75, d4 = (i - t * 75) * 4;
    float4 v = *(const float4*)(embed + (long)ctx_ids[m * 50 + t] * DD + d4);
    X[t][d4 + 0] = (_Float16)v.x; X[t][d4 + 1] = (_Float16)v.y;
    X[t][d4 + 2] = (_Float16)v.z; X[t][d4 + 3] = (_Float16)v.w;
  }
  __syncthreads();
  v8f c[3] = {};
  int fbase = fg * 128 + wave * 16;
  for (int k = 0; k < 5; ++k)
    for (int ks = 0; ks < 10; ++ks) {
      v16h bt = ldBt(&Wt[((long)k * FF + fbase) * 320 + ks * 32], 320);
      for (int tt = 0; tt < 3; ++tt) {
        v16h a = ldA(&X[tt * 16 + k][ks * 32], 320);
        c[tt] = wmma_f16(a, bt, c[tt]);
      }
    }
  int nn = lane & 15, mb = (lane & 16) ? 8 : 0;
  int f = fbase + nn;
  float bias = b[f];
  float acc = 0.f;
#pragma unroll
  for (int tt = 0; tt < 3; ++tt)
#pragma unroll
    for (int r = 0; r < 8; ++r) {
      int t = tt * 16 + mb + r;
      if (t < 46) acc += fmaxf(c[tt][r] + bias, 0.f);
    }
  acc += __shfl_xor(acc, 16, 32);
  if (lane < 16) mc[m * FF + f] = acc * (1.f / 46.f);
}

// l2-normalize rows: blocks 0..63 ms, 64..127 mc, 128 bert->mdn
__global__ __launch_bounds__(256) void k_normalize(float* ms, float* mc, const float* bert, float* mdn) {
  __shared__ float red[256];
  int bidx = blockIdx.x;
  float* dst; const float* src;
  if (bidx < 64)       { dst = ms + bidx * FF;        src = dst; }
  else if (bidx < 128) { dst = mc + (bidx - 64) * FF; src = dst; }
  else                 { dst = mdn;                   src = bert; }
  float ss = 0.f;
  for (int i = threadIdx.x; i < FF; i += 256) { float v = src[i]; ss += v * v; }
  float nrm = sqrtf(bsum(ss, red));
  float inv = 1.f / fmaxf(nrm, 1e-6f);
  for (int i = threadIdx.x; i < FF; i += 256) dst[i] = src[i] * inv;
}

// gather candidate embeddings as f16, row-major [512][320] (N-major for WMMA B)
__global__ void k_cand_gather(const float* embed, const int* cand_ids, _Float16* candh) {
  int idx = blockIdx.x * 256 + threadIdx.x;
  if (idx >= NN * 80) return;
  int n = idx / 80, d4 = (idx - n * 80) * 4;
  _Float16* p = candh + (long)n * 320 + d4;
  if (d4 < DD) {
    float4 v = *(const float4*)(embed + (long)cand_ids[n] * DD + d4);
    p[0] = (_Float16)v.x; p[1] = (_Float16)v.y; p[2] = (_Float16)v.z; p[3] = (_Float16)v.w;
  } else {
    p[0] = (_Float16)0.f; p[1] = (_Float16)0.f; p[2] = (_Float16)0.f; p[3] = (_Float16)0.f;
  }
}

// cosine features: block per candidate n
__global__ __launch_bounds__(256) void k_cos(const float* title, const float* msn, const float* mcn,
                                             const float* mdn, float* cst, float* cct, float* cdt) {
  __shared__ float tn[FF];
  __shared__ float red[256];
  int n = blockIdx.x, tid = threadIdx.x;
  float ss = 0.f;
  for (int i = tid; i < FF; i += 256) { float v = title[(long)n * FF + i]; tn[i] = v; ss += v * v; }
  float nrm = sqrtf(bsum(ss, red));
  float inv = 1.f / fmaxf(nrm, 1e-6f);
  for (int i = tid; i < FF; i += 256) tn[i] *= inv;
  __syncthreads();
  for (int w = tid; w < 129; w += 256) {
    const float* vec = (w < 64) ? (msn + w * FF) : (w < 128) ? (mcn + (w - 64) * FF) : mdn;
    float dsum = 0.f;
    for (int i = 0; i < FF; ++i) dsum += vec[i] * tn[i];
    if (w < 64) cst[w * NN + n] = dsum;
    else if (w < 128) cct[(w - 64) * NN + n] = dsum;
    else cdt[n] = dsum;
  }
}

// att_db: per (n,h): S=QK^T/sqrt(hd), softmax over k, O=P@K, max over q. q chunked by 32.
// Kh kept in both [k][d] (S-phase B) and [d][k] (O-phase B) layouts so all WMMA operands
// are per-lane contiguous in LDS.
__global__ __launch_bounds__(256) void k_att_db(const float* embed, const int* doc_ids,
                                                const int* body_ids, float* dbp) {
  __shared__ _Float16 Kh[224][64];   // 28KB  [k][d]
  __shared__ _Float16 KhT[64][224];  // 28KB  [d][k]
  __shared__ _Float16 Qh[32][64];    // 4KB
  __shared__ _Float16 Sm[32][224];   // 14KB
  int n = blockIdx.x, h = blockIdx.y, tid = threadIdx.x;
  int wave = tid >> 5, lane = tid & 31;
  for (int i = tid; i < 224 * 64; i += 256) ((_Float16*)Kh)[i] = (_Float16)0.f;
  for (int i = tid; i < 64 * 224; i += 256) ((_Float16*)KhT)[i] = (_Float16)0.f;
  __syncthreads();
  for (int i = tid; i < BODY * 15; i += 256) {
    int k = i / 15, d4 = (i - k * 15) * 4;
    float4 v = *(const float4*)(embed + (long)body_ids[n * BODY + k] * DD + h * HD + d4);
    _Float16 h0 = (_Float16)v.x, h1 = (_Float16)v.y, h2 = (_Float16)v.z, h3 = (_Float16)v.w;
    Kh[k][d4 + 0] = h0; Kh[k][d4 + 1] = h1; Kh[k][d4 + 2] = h2; Kh[k][d4 + 3] = h3;
    KhT[d4 + 0][k] = h0; KhT[d4 + 1][k] = h1; KhT[d4 + 2][k] = h2; KhT[d4 + 3][k] = h3;
  }
  const float sc = 0.129099445f;  // 1/sqrt(60)
  for (int chunk = 0; chunk < 4; ++chunk) {
    int q0 = chunk * 32;
    __syncthreads();
    for (int i = tid; i < 32 * 64; i += 256) ((_Float16*)Qh)[i] = (_Float16)0.f;
    __syncthreads();
    for (int i = tid; i < 32 * 15; i += 256) {
      int q = i / 15, d4 = (i - q * 15) * 4;
      if (q0 + q < DOC) {
        float4 v = *(const float4*)(embed + (long)doc_ids[q0 + q] * DD + h * HD + d4);
        Qh[q][d4 + 0] = (_Float16)v.x; Qh[q][d4 + 1] = (_Float16)v.y;
        Qh[q][d4 + 2] = (_Float16)v.z; Qh[q][d4 + 3] = (_Float16)v.w;
      }
    }
    __syncthreads();
    for (int tile = wave; tile < 28; tile += 8) {   // 2 qtiles x 14 ktiles
      int qt = tile & 1, kt = tile >> 1;
      v8f c = {};
      for (int ks = 0; ks < 2; ++ks)
        c = wmma_f16(ldA(&Qh[qt * 16][ks * 32], 64), ldBt(&Kh[kt * 16][ks * 32], 64), c);
      stD16(&Sm[qt * 16][kt * 16], 224, c, sc);
    }
    __syncthreads();
    for (int r = tid; r < 32; r += 256) {
      if (q0 + r < DOC) {
        float mx = -1e30f;
        for (int k = 0; k < BODY; ++k) mx = fmaxf(mx, (float)Sm[r][k]);
        float ssum = 0.f;
        for (int k = 0; k < BODY; ++k) { float e = __expf((float)Sm[r][k] - mx); ssum += e; Sm[r][k] = (_Float16)e; }
        float inv = 1.f / ssum;
        for (int k = 0; k < BODY; ++k) Sm[r][k] = (_Float16)((float)Sm[r][k] * inv);
        for (int k = BODY; k < 224; ++k) Sm[r][k] = (_Float16)0.f;
      } else {
        for (int k = 0; k < 224; ++k) Sm[r][k] = (_Float16)0.f;
      }
    }
    __syncthreads();
    float* maxbuf = (float*)&Qh[0][0];   // reuse Qh: [2][64] f32
    for (int tile = wave; tile < 8; tile += 8) {    // 2 qtiles x 4 dtiles
      int qt = tile >> 2, dt = tile & 3;
      v8f c = {};
      for (int ks = 0; ks < 7; ++ks)
        c = wmma_f16(ldA(&Sm[qt * 16][ks * 32], 224), ldBt(&KhT[dt * 16][ks * 32], 224), c);
      int nn = lane & 15, mb = (lane & 16) ? 8 : 0;
      float mx = -1e30f;
#pragma unroll
      for (int r = 0; r < 8; ++r) { int q = q0 + qt * 16 + mb + r; if (q < DOC) mx = fmaxf(mx, c[r]); }
      mx = fmaxf(mx, __shfl_xor(mx, 16, 32));
      if (lane < 16) maxbuf[qt * 64 + dt * 16 + nn] = mx;
    }
    __syncthreads();
    for (int d = tid; d < HD; d += 256) {
      float v = fmaxf(maxbuf[d], maxbuf[64 + d]);
      float* dst = &dbp[n * DD + h * HD + d];
      *dst = (chunk == 0) ? v : fmaxf(*dst, v);
    }
  }
}

// att_bd: per (n,h): S^T=K@Q^T/sqrt(hd), softmax over q, O=P2@Q, max over k. k halved.
__global__ __launch_bounds__(256) void k_att_bd(const float* embed, const int* doc_ids,
                                                const int* body_ids, float* bdp) {
  __shared__ _Float16 Qh[128][64];   // 16KB  [q][d]
  __shared__ _Float16 QhT[64][128];  // 16KB  [d][q]
  __shared__ _Float16 Kh[112][64];   // 14KB
  __shared__ _Float16 St[112][128];  // 28KB
  int n = blockIdx.x, h = blockIdx.y, tid = threadIdx.x;
  int wave = tid >> 5, lane = tid & 31;
  for (int i = tid; i < 128 * 64; i += 256) ((_Float16*)Qh)[i] = (_Float16)0.f;
  for (int i = tid; i < 64 * 128; i += 256) ((_Float16*)QhT)[i] = (_Float16)0.f;
  __syncthreads();
  for (int i = tid; i < DOC * 15; i += 256) {
    int q = i / 15, d4 = (i - q * 15) * 4;
    float4 v = *(const float4*)(embed + (long)doc_ids[q] * DD + h * HD + d4);
    _Float16 h0 = (_Float16)v.x, h1 = (_Float16)v.y, h2 = (_Float16)v.z, h3 = (_Float16)v.w;
    Qh[q][d4 + 0] = h0; Qh[q][d4 + 1] = h1; Qh[q][d4 + 2] = h2; Qh[q][d4 + 3] = h3;
    QhT[d4 + 0][q] = h0; QhT[d4 + 1][q] = h1; QhT[d4 + 2][q] = h2; QhT[d4 + 3][q] = h3;
  }
  const float sc = 0.129099445f;
  for (int half = 0; half < 2; ++half) {
    int k0 = half * 112;
    __syncthreads();
    for (int i = tid; i < 112 * 64; i += 256) ((_Float16*)Kh)[i] = (_Float16)0.f;
    __syncthreads();
    for (int i = tid; i < 112 * 15; i += 256) {
      int k = i / 15, d4 = (i - k * 15) * 4;
      if (k0 + k < BODY) {
        float4 v = *(const float4*)(embed + (long)body_ids[n * BODY + k0 + k] * DD + h * HD + d4);
        Kh[k][d4 + 0] = (_Float16)v.x; Kh[k][d4 + 1] = (_Float16)v.y;
        Kh[k][d4 + 2] = (_Float16)v.z; Kh[k][d4 + 3] = (_Float16)v.w;
      }
    }
    __syncthreads();
    for (int tile = wave; tile < 56; tile += 8) {   // 7 ktiles x 8 qtiles
      int kt = tile >> 3, qt = tile & 7;
      v8f c = {};
      for (int ks = 0; ks < 2; ++ks)
        c = wmma_f16(ldA(&Kh[kt * 16][ks * 32], 64), ldBt(&Qh[qt * 16][ks * 32], 64), c);
      stD16(&St[kt * 16][qt * 16], 128, c, sc);
    }
    __syncthreads();
    for (int r = tid; r < 112; r += 256) {
      if (k0 + r < BODY) {
        float mx = -1e30f;
        for (int q = 0; q < DOC; ++q) mx = fmaxf(mx, (float)St[r][q]);
        float ssum = 0.f;
        for (int q = 0; q < DOC; ++q) { float e = __expf((float)St[r][q] - mx); ssum += e; St[r][q] = (_Float16)e; }
        float inv = 1.f / ssum;
        for (int q = 0; q < DOC; ++q) St[r][q] = (_Float16)((float)St[r][q] * inv);
        for (int q = DOC; q < 128; ++q) St[r][q] = (_Float16)0.f;
      } else {
        for (int q = 0; q < 128; ++q) St[r][q] = (_Float16)0.f;
      }
    }
    __syncthreads();
    float* maxbuf = (float*)&Kh[0][0];   // reuse Kh: [7][64] f32
    for (int tile = wave; tile < 28; tile += 8) {   // 7 ktiles x 4 dtiles
      int kt = tile >> 2, dt = tile & 3;
      v8f c = {};
      for (int ks = 0; ks < 4; ++ks)
        c = wmma_f16(ldA(&St[kt * 16][ks * 32], 128), ldBt(&QhT[dt * 16][ks * 32], 128), c);
      int nn = lane & 15, mb = (lane & 16) ? 8 : 0;
      float mx = -1e30f;
#pragma unroll
      for (int r = 0; r < 8; ++r) { int k = k0 + kt * 16 + mb + r; if (k < BODY) mx = fmaxf(mx, c[r]); }
      mx = fmaxf(mx, __shfl_xor(mx, 16, 32));
      if (lane < 16) maxbuf[kt * 64 + dt * 16 + nn] = mx;
    }
    __syncthreads();
    for (int d = tid; d < HD; d += 256) {
      float v = -1e30f;
      for (int kt = 0; kt < 7; ++kt) v = fmaxf(v, maxbuf[kt * 64 + d]);
      float* dst = &bdp[n * DD + h * HD + d];
      *dst = (half == 0) ? v : fmaxf(*dst, v);
    }
  }
}

// att_score MLP: sigmoid(relu(comb@W1+b1)@W2+b2), block per candidate
__global__ __launch_bounds__(256) void k_att_score(const float* dbp, const float* bdp,
                                                   const float* W1, const float* b1,
                                                   const float* W2, const float* b2, float* as) {
  __shared__ float comb[2 * DD];
  __shared__ float red[256];
  int n = blockIdx.x, tid = threadIdx.x;
  for (int i = tid; i < 2 * DD; i += 256)
    comb[i] = (i < DD) ? dbp[n * DD + i] : bdp[n * DD + (i - DD)];
  __syncthreads();
  float acc = 0.f;
  for (int j = tid; j < DD; j += 256) {
    float hsum = b1[j];
    for (int i = 0; i < 2 * DD; ++i) hsum += comb[i] * W1[i * DD + j];
    acc += fmaxf(hsum, 0.f) * W2[j];
  }
  float tot = bsum(acc, red);
  if (tid == 0) as[n] = 1.f / (1.f + __expf(-(tot + b2[0])));
}

// local_att[m,n] = max_t dot(ctx[m,t], cand[n]) / sqrt(D). Block per mention, WMMA.
// candh is [n][d] (N-major) so B loads are contiguous global reads.
__global__ __launch_bounds__(256) void k_local_att(const float* embed, const int* ctx_ids,
                                                   const _Float16* candh, float* latt) {
  __shared__ _Float16 X[64][320];    // 40KB
  int m = blockIdx.x, tid = threadIdx.x;
  int wave = tid >> 5, lane = tid & 31;
  for (int i = tid; i < 64 * 320; i += 256) ((_Float16*)X)[i] = (_Float16)0.f;
  __syncthreads();
  for (int i = tid; i < 50 * 75; i += 256) {
    int t = i / 75, d4 = (i - t * 75) * 4;
    float4 v = *(const float4*)(embed + (long)ctx_ids[m * 50 + t] * DD + d4);
    X[t][d4 + 0] = (_Float16)v.x; X[t][d4 + 1] = (_Float16)v.y;
    X[t][d4 + 2] = (_Float16)v.z; X[t][d4 + 3] = (_Float16)v.w;
  }
  __syncthreads();
  for (int nt = wave; nt < 32; nt += 8) {
    v8f c[4] = {};
    for (int ks = 0; ks < 10; ++ks) {
      v16h bt = ldBt(&candh[(long)(nt * 16) * 320 + ks * 32], 320);
      for (int tt = 0; tt < 4; ++tt) {
        v16h a = ldA(&X[tt * 16][ks * 32], 320);
        c[tt] = wmma_f16(a, bt, c[tt]);
      }
    }
    int nn = lane & 15, mb = (lane & 16) ? 8 : 0;
    float mx = -1e30f;
#pragma unroll
    for (int tt = 0; tt < 4; ++tt)
#pragma unroll
      for (int r = 0; r < 8; ++r) { int t = tt * 16 + mb + r; if (t < 50) mx = fmaxf(mx, c[tt][r]); }
    mx = fmaxf(mx, __shfl_xor(mx, 16, 32));
    if (lane < 16) latt[m * NN + nt * 16 + nn] = mx * 0.0577350269f;  // 1/sqrt(300)
  }
}

// ranker: raw features -> masked z-score -> masked softmax (sm). Block per mention.
__global__ __launch_bounds__(256) void k_rank(const float* hand, const float* m2c, const float* latt,
                                              const float* as, const float* cst, const float* cdt,
                                              const float* cct, const float* llW, const float* llb,
                                              const float* mask, float* sm) {
  __shared__ float raw_s[NN], msk[NN];
  __shared__ float red[256];
  int m = blockIdx.x, tid = threadIdx.x;
  float w0 = llW[0], w1 = llW[1], w2 = llW[2], w3 = llW[3], w4 = llW[4], w5 = llW[5], w6 = llW[6];
  float bb = llb[0];
  for (int n = tid; n < NN; n += 256) {
    raw_s[n] = w0 * hand[n] + w1 * m2c[n] + w2 * latt[m * NN + n] + w3 * as[n]
             + w4 * cst[m * NN + n] + w5 * cdt[n] + w6 * cct[m * NN + n] + bb;
    msk[n] = mask[m * NN + n];
  }
  __syncthreads();
  float c = 0.f, su = 0.f;
  for (int n = tid; n < NN; n += 256) { c += msk[n]; su += raw_s[n] * msk[n]; }
  float cnt = bsum(c, red);
  float mean = bsum(su, red) / cnt;
  float vv = 0.f;
  for (int n = tid; n < NN; n += 256) { float d = raw_s[n] - mean; vv += d * d * msk[n]; }
  float var = bsum(vv, red) / (cnt - 1.f);
  float istd = 1.f / sqrtf(var);
  float mx = -1e30f;
  for (int n = tid; n < NN; n += 256) if (msk[n] > 0.f) mx = fmaxf(mx, (raw_s[n] - mean) * istd);
  mx = bmax(mx, red);
  float se = 0.f;
  for (int n = tid; n < NN; n += 256) if (msk[n] > 0.f) se += __expf((raw_s[n] - mean) * istd - mx);
  se = bsum(se, red);
  for (int n = tid; n < NN; n += 256) {
    float z = (raw_s[n] - mean) * istd;
    sm[m * NN + n] = (msk[n] > 0.f) ? __expf(z - mx) / se * msk[n] : 0.f;
  }
}

__global__ void k_s(const float* sm, float* s) {
  int n = blockIdx.x * 256 + threadIdx.x;
  if (n >= NN) return;
  float acc = 0.f;
  for (int m = 0; m < MM; ++m) acc += sm[m * NN + n];
  s[n] = acc;
}

// top-3 per mention -> keep mask (benign same-value writes)
__global__ __launch_bounds__(256) void k_top3(const float* sm, float* keep) {
  __shared__ float vals[NN];
  __shared__ float redv[256];
  __shared__ int   redi[256];
  int m = blockIdx.x, tid = threadIdx.x;
  for (int n = tid; n < NN; n += 256) vals[n] = sm[m * NN + n];
  __syncthreads();
  for (int pick = 0; pick < 3; ++pick) {
    float bv = -1e30f; int bi = 0x7fffffff;
    for (int n = tid; n < NN; n += 256) {
      float v = vals[n];
      if (v > bv || (v == bv && n < bi)) { bv = v; bi = n; }
    }
    redv[tid] = bv; redi[tid] = bi; __syncthreads();
    for (int s = 128; s > 0; s >>= 1) {
      if (tid < s) {
        if (redv[tid + s] > redv[tid] ||
            (redv[tid + s] == redv[tid] && redi[tid + s] < redi[tid])) {
          redv[tid] = redv[tid + s]; redi[tid] = redi[tid + s];
        }
      }
      __syncthreads();
    }
    int w = redi[0];
    __syncthreads();
    if (tid == 0) { keep[w] = 1.f; vals[w] = -2e30f; }
    __syncthreads();
  }
}

// SR = esr*eee*keep, row-normalized
__global__ __launch_bounds__(256) void k_sr(const float* esr, const float* eee,
                                            const float* keep, float* SRn) {
  __shared__ float red[256];
  int i = blockIdx.x, tid = threadIdx.x;
  float acc = 0.f;
  for (int j = tid; j < NN; j += 256) acc += esr[(long)i * NN + j] * eee[(long)i * NN + j] * keep[j];
  float rs = bsum(acc, red);
  float inv = (fabsf(rs) < 1e-6f) ? 1.f : 1.f / rs;
  for (int j = tid; j < NN; j += 256)
    SRn[(long)i * NN + j] = esr[(long)i * NN + j] * eee[(long)i * NN + j] * keep[j] * inv;
}

// power iteration (single block, sequential, deterministic)
__global__ __launch_bounds__(256) void k_fai(const float* s, const float* SRn, const float* lamda,
                                             const int* rk, float* fai) {
  __shared__ float f[NN], fn[NN];
  int tid = threadIdx.x;
  float lam = lamda[0];
  int iters = rk[0];
  for (int n = tid; n < NN; n += 256) f[n] = s[n];
  __syncthreads();
  for (int it = 0; it < iters; ++it) {
    for (int n = tid; n < NN; n += 256) {
      float acc = 0.f;
      for (int i = 0; i < NN; ++i) acc += f[i] * SRn[(long)i * NN + n];
      fn[n] = (1.f - lam) * acc + lam * s[n];
    }
    __syncthreads();
    for (int n = tid; n < NN; n += 256) f[n] = fn[n];
    __syncthreads();
  }
  for (int n = tid; n < NN; n += 256) fai[n] = f[n];
}

__global__ void k_out(const float* s, const float* fai, const float* mask, float* out) {
  int idx = blockIdx.x * 256 + threadIdx.x;
  if (idx < NN) out[idx] = s[idx];
  else if (idx < 2 * NN) out[idx] = fai[idx - NN];
  else if (idx < 2 * NN + MM * NN) {
    int t = idx - 2 * NN;
    out[idx] = mask[t] * fai[t % NN];
  }
}

extern "C" void kernel_launch(void* const* d_in, const int* in_sizes, int n_in,
                              void* d_out, int out_size, void* d_ws, size_t ws_size,
                              hipStream_t stream) {
  (void)in_sizes; (void)n_in; (void)out_size; (void)ws_size;
  const float* embed      = (const float*)d_in[0];
  const float* W_ms       = (const float*)d_in[1];
  const float* b_ms       = (const float*)d_in[2];
  const float* W_mc       = (const float*)d_in[3];
  const float* b_mc       = (const float*)d_in[4];
  const float* att_W1     = (const float*)d_in[5];
  const float* att_b1     = (const float*)d_in[6];
  const float* att_W2     = (const float*)d_in[7];
  const float* att_b2     = (const float*)d_in[8];
  const float* ll_W       = (const float*)d_in[9];
  const float* ll_b       = (const float*)d_in[10];
  const float* lamda      = (const float*)d_in[11];
  const int*   men_ids    = (const int*)d_in[12];
  const int*   ctx_ids    = (const int*)d_in[13];
  const int*   doc_ids    = (const int*)d_in[14];
  const int*   body_ids   = (const int*)d_in[15];
  const int*   cand_ids   = (const int*)d_in[16];
  const float* title_vec  = (const float*)d_in[17];
  const float* bert_doc   = (const float*)d_in[18];
  const float* hand       = (const float*)d_in[19];
  const float* m2c        = (const float*)d_in[20];
  const float* ment_ent   = (const float*)d_in[21];
  const float* ent_ent    = (const float*)d_in[22];
  const float* ent_sr     = (const float*)d_in[23];
  const int*   random_k   = (const int*)d_in[24];
  float* out = (float*)d_out;

  char* w = (char*)d_ws;
  auto alloc = [&](size_t bytes) { char* p = w; w += (bytes + 255) & ~(size_t)255; return p; };
  float*    ms    = (float*)alloc(MM * FF * 4);
  float*    mc    = (float*)alloc(MM * FF * 4);
  float*    mdn   = (float*)alloc(FF * 4);
  float*    cst   = (float*)alloc(MM * NN * 4);
  float*    cct   = (float*)alloc(MM * NN * 4);
  float*    cdt   = (float*)alloc(NN * 4);
  float*    dbp   = (float*)alloc(NN * DD * 4);
  float*    bdp   = (float*)alloc(NN * DD * 4);
  float*    as    = (float*)alloc(NN * 4);
  float*    latt  = (float*)alloc(MM * NN * 4);
  float*    sm    = (float*)alloc(MM * NN * 4);
  float*    s     = (float*)alloc(NN * 4);
  float*    keep  = (float*)alloc(NN * 4);
  float*    SRn   = (float*)alloc((size_t)NN * NN * 4);
  float*    fai   = (float*)alloc(NN * 4);
  _Float16* candh = (_Float16*)alloc((size_t)NN * 320 * 2);
  _Float16* Wmct  = (_Float16*)alloc((size_t)5 * FF * 320 * 2);

  hipLaunchKernelGGL(k_zero, dim3(2), dim3(256), 0, stream, keep);
  hipLaunchKernelGGL(k_wmc_t, dim3(4800), dim3(256), 0, stream, W_mc, Wmct);
  hipLaunchKernelGGL(k_conv_ms, dim3(MM), dim3(256), 0, stream, embed, men_ids, W_ms, b_ms, ms);
  hipLaunchKernelGGL(k_conv_mc, dim3(6, MM), dim3(256), 0, stream, embed, ctx_ids, Wmct, b_mc, mc);
  hipLaunchKernelGGL(k_normalize, dim3(129), dim3(256), 0, stream, ms, mc, bert_doc, mdn);
  hipLaunchKernelGGL(k_cand_gather, dim3(160), dim3(256), 0, stream, embed, cand_ids, candh);
  hipLaunchKernelGGL(k_cos, dim3(NN), dim3(256), 0, stream, title_vec, ms, mc, mdn, cst, cct, cdt);
  hipLaunchKernelGGL(k_att_db, dim3(NN, HH), dim3(256), 0, stream, embed, doc_ids, body_ids, dbp);
  hipLaunchKernelGGL(k_att_bd, dim3(NN, HH), dim3(256), 0, stream, embed, doc_ids, body_ids, bdp);
  hipLaunchKernelGGL(k_att_score, dim3(NN), dim3(256), 0, stream, dbp, bdp, att_W1, att_b1, att_W2, att_b2, as);
  hipLaunchKernelGGL(k_local_att, dim3(MM), dim3(256), 0, stream, embed, ctx_ids, candh, latt);
  hipLaunchKernelGGL(k_rank, dim3(MM), dim3(256), 0, stream, hand, m2c, latt, as, cst, cdt, cct, ll_W, ll_b, ment_ent, sm);
  hipLaunchKernelGGL(k_s, dim3(2), dim3(256), 0, stream, sm, s);
  hipLaunchKernelGGL(k_top3, dim3(MM), dim3(256), 0, stream, sm, keep);
  hipLaunchKernelGGL(k_sr, dim3(NN), dim3(256), 0, stream, ent_sr, ent_ent, keep, SRn);
  hipLaunchKernelGGL(k_fai, dim3(1), dim3(256), 0, stream, s, SRn, lamda, random_k, fai);
  hipLaunchKernelGGL(k_out, dim3(132), dim3(256), 0, stream, s, fai, ment_ent, out);
}
